// DSA_11192684773671
// MI455X (gfx1250) — matrix operand order repaired and checked by reading, not compiled, not run on hardware
//
#include <hip/hip_runtime.h>

typedef __bf16 bf16;
typedef __attribute__((ext_vector_type(16))) __bf16 v16bf;
typedef __attribute__((ext_vector_type(8)))  __bf16 v8bf;
typedef __attribute__((ext_vector_type(4)))  __bf16 v4bf;
typedef __attribute__((ext_vector_type(8)))  float  v8f;

constexpr int kB  = 4;
constexpr int kT  = 5;
constexpr int kC  = 512;
constexpr int kHW = 1024;          // 32x32
constexpr int kN  = kB * kHW;      // 4096 conv GEMM N
constexpr long kFrame = (long)kC * kHW;          // elements per (b) frame slice
constexpr long kFeatB = (long)kT * kC * kHW;     // batch stride in features

constexpr int LDK = 72;            // LDS k-stride (pad: 144B rows, 16B aligned)

__device__ __forceinline__ bf16 f2b(float x) { return (bf16)x; }

// ---------------------------------------------------------------------------
// Wave-level 32x32 WMMA step over a 64xBK A tile and BKx128 B tile (both in
// fragment-friendly LDS layouts).  kb selects the 32-wide k-slice.
// A fragment (16-bit 16x32): lanes0-15 M=lane, K {0..7,16..23};
// lanes16-31 K {8..15,24..31}.  B/C/D: rows striped across lanes.
// ---------------------------------------------------------------------------
__device__ __forceinline__ void wave_mma_k32(const bf16 (*As)[LDK], const bf16 (*BsT)[LDK],
                                             v8f acc[2][2], int wm, int wn, int lane, int kb)
{
    const int lm = lane & 15;
    const int hi = lane >> 4;
    v16bf a[2], b[2];
#pragma unroll
    for (int s = 0; s < 2; ++s) {
        const bf16* pa = &As[wm + s * 16 + lm][kb + hi * 8];
        v8bf lo = *(const v8bf*)pa;          // K 0..7   (or 8..15)
        v8bf hh = *(const v8bf*)(pa + 16);   // K 16..23 (or 24..31)
        a[s] = __builtin_shufflevector(lo, hh, 0, 1, 2, 3, 4, 5, 6, 7,
                                               8, 9, 10, 11, 12, 13, 14, 15);
    }
#pragma unroll
    for (int s = 0; s < 2; ++s) {
        const bf16* pb = &BsT[wn + s * 16 + lm][kb + hi * 8];
        v8bf b0 = *(const v8bf*)pb;
        v8bf b1 = *(const v8bf*)(pb + 16);
        b[s] = __builtin_shufflevector(b0, b1, 0, 1, 2, 3, 4, 5, 6, 7,
                                               8, 9, 10, 11, 12, 13, 14, 15);
    }
#pragma unroll
    for (int i = 0; i < 2; ++i)
#pragma unroll
        for (int j = 0; j < 2; ++j)
            acc[i][j] = __builtin_amdgcn_wmma_f32_16x16x32_bf16(
                false, a[i], false, b[j], (short)0, acc[i][j], false, false);
}

// ---------------------------------------------------------------------------
// Batched GEMM, bf16 in / f32 out.  TA=true: A stored (K,M) row-major (TN);
// TA=false: A stored (M,K) row-major (NN).  B stored (K,N) row-major.
// Block tile 64x128x64, 8 waves (2x4), double-buffered LDS pipeline:
// the fill for tile i+1 issues before the wmma of tile i (one barrier/iter).
// ---------------------------------------------------------------------------
template <bool TA>
__global__ __launch_bounds__(256) void gemm_bf16(
    const bf16* __restrict__ A, const bf16* __restrict__ Bm, float* __restrict__ Cm,
    int M, int N, int K, long sA, long sB, long sC)
{
    __shared__ bf16 As[2][64][LDK];
    __shared__ bf16 BsT[2][128][LDK];     // [n][k]
    const int bz = blockIdx.z;
    A  += (long)bz * sA;
    Bm += (long)bz * sB;
    Cm += (long)bz * sC;
    const int m0 = blockIdx.x * 64, n0 = blockIdx.y * 128;
    const int tid = threadIdx.x, lane = tid & 31, wid = tid >> 5;
    const int wm = (wid >> 2) * 32, wn = (wid & 3) * 32;
    v8f acc[2][2] = {};

    const int a_m_tn = tid & 63,  a_kq_tn = (tid >> 6) * 16;   // TN: 16 k per thread
    const int a_m_nn = tid >> 2,  a_kc_nn = (tid & 3) * 16;    // NN: 16 k per thread
    const int b_n = tid & 127,    b_kh = (tid >> 7) * 32;      // 32 k per thread

    auto fill = [&](int buf, int k0) {
        if (TA) {   // A (K,M): lane-coalesced u16 loads, vector LDS stores
            v8bf t0, t1;
#pragma unroll
            for (int e = 0; e < 8; ++e) {
                t0[e] = A[(long)(k0 + a_kq_tn + e) * M + (m0 + a_m_tn)];
                t1[e] = A[(long)(k0 + a_kq_tn + 8 + e) * M + (m0 + a_m_tn)];
            }
            *(v8bf*)&As[buf][a_m_tn][a_kq_tn]     = t0;
            *(v8bf*)&As[buf][a_m_tn][a_kq_tn + 8] = t1;
        } else {    // A (M,K): fully vectorized both sides
            const bf16* g = &A[(long)(m0 + a_m_nn) * K + k0 + a_kc_nn];
            *(v8bf*)&As[buf][a_m_nn][a_kc_nn]     = *(const v8bf*)g;
            *(v8bf*)&As[buf][a_m_nn][a_kc_nn + 8] = *(const v8bf*)(g + 8);
        }
#pragma unroll
        for (int q = 0; q < 4; ++q) {
            v8bf t;
#pragma unroll
            for (int e = 0; e < 8; ++e)
                t[e] = Bm[(long)(k0 + b_kh + q * 8 + e) * N + (n0 + b_n)];
            *(v8bf*)&BsT[buf][b_n][b_kh + q * 8] = t;
        }
    };

    const int nK = K >> 6;
    fill(0, 0);
    __syncthreads();
    for (int kt = 0; kt < nK; ++kt) {
        const int cur = kt & 1;
        if (kt + 1 < nK) fill(cur ^ 1, (kt + 1) << 6);
        wave_mma_k32(As[cur], BsT[cur], acc, wm, wn, lane, 0);
        wave_mma_k32(As[cur], BsT[cur], acc, wm, wn, lane, 32);
        __syncthreads();
    }

    const int lm = lane & 15, hi = lane >> 4;
#pragma unroll
    for (int i = 0; i < 2; ++i)
#pragma unroll
        for (int j = 0; j < 2; ++j)
#pragma unroll
            for (int r = 0; r < 8; ++r) {
                const int row = m0 + wm + i * 16 + hi * 8 + r;
                const int col = n0 + wn + j * 16 + lm;
                Cm[(long)row * N + col] = acc[i][j][r];
            }
}

// ---------------------------------------------------------------------------
// Implicit-GEMM 3x3 conv (pad 1) with channel-concat input (up to 3 sources,
// 512 channels each), optional input relu, per-channel alpha/beta epilogue
// (BN+bias folded), optional residual add and output relu.
// M=Cout=512, N=4096 (b*HW+hw), K=Cin*9.  Double-buffered like gemm_bf16.
// ---------------------------------------------------------------------------
__global__ __launch_bounds__(256) void conv_gemm_bf16(
    const bf16* __restrict__ W,                 // (512, Cin*9)
    const float* __restrict__ x0, long st0,
    const float* __restrict__ x1, long st1,
    const float* __restrict__ x2, long st2,
    int Cin, int reluIn,
    float* __restrict__ out, long stOut,
    const float* __restrict__ alpha, const float* __restrict__ beta,
    const float* __restrict__ resid, long stRes, int reluOut)
{
    __shared__ bf16 As[2][64][LDK];
    __shared__ bf16 BsT[2][128][LDK];     // [n][k]
    const int K = Cin * 9;
    const int m0 = blockIdx.x * 64, n0 = blockIdx.y * 128;
    const int tid = threadIdx.x, lane = tid & 31, wid = tid >> 5;
    const int wm = (wid >> 2) * 32, wn = (wid & 3) * 32;
    v8f acc[2][2] = {};

    const int a_m = tid >> 2, a_kc = (tid & 3) * 16;
    const int b_n = tid & 127, b_kh = (tid >> 7) * 32;
    // fixed spatial decode for this thread's im2col column
    const int nn = n0 + b_n;
    const int b  = nn >> 10, hwb = nn & 1023;
    const int hb = hwb >> 5, wb = hwb & 31;

    auto fill = [&](int buf, int k0) {
        // weights (M,K) row-major: vectorized both sides
        const bf16* g = &W[(long)(m0 + a_m) * K + k0 + a_kc];
        *(v8bf*)&As[buf][a_m][a_kc]     = *(const v8bf*)g;
        *(v8bf*)&As[buf][a_m][a_kc + 8] = *(const v8bf*)(g + 8);

        // implicit im2col column: incremental (ci, tap) walk over 32 k's
        int kk = k0 + b_kh;
        int ci = kk / 9;
        int rr = kk - ci * 9;
#pragma unroll
        for (int q = 0; q < 4; ++q) {
            v8bf t;
#pragma unroll
            for (int e = 0; e < 8; ++e) {
                const int dh = rr / 3 - 1;
                const int dw = rr - (rr / 3) * 3 - 1;
                const int h = hb + dh, w = wb + dw;
                float v = 0.f;
                if ((unsigned)h < 32u && (unsigned)w < 32u) {
                    const float* src; long st; int cl = ci;
                    if (cl < 512)       { src = x0; st = st0; }
                    else if (cl < 1024) { src = x1; st = st1; cl -= 512; }
                    else                { src = x2; st = st2; cl -= 1024; }
                    v = src[(long)b * st + (long)cl * kHW + h * 32 + w];
                    if (reluIn) v = fmaxf(v, 0.f);
                }
                t[e] = f2b(v);
                if (++rr == 9) { rr = 0; ++ci; }
            }
            *(v8bf*)&BsT[buf][b_n][b_kh + q * 8] = t;
        }
    };

    const int nK = K >> 6;
    fill(0, 0);
    __syncthreads();
    for (int kt = 0; kt < nK; ++kt) {
        const int cur = kt & 1;
        if (kt + 1 < nK) fill(cur ^ 1, (kt + 1) << 6);
        wave_mma_k32(As[cur], BsT[cur], acc, wm, wn, lane, 0);
        wave_mma_k32(As[cur], BsT[cur], acc, wm, wn, lane, 32);
        __syncthreads();
    }

    const int lm = lane & 15, hi = lane >> 4;
#pragma unroll
    for (int i = 0; i < 2; ++i)
#pragma unroll
        for (int j = 0; j < 2; ++j)
#pragma unroll
            for (int r = 0; r < 8; ++r) {
                const int row = m0 + wm + i * 16 + hi * 8 + r;
                const int col = n0 + wn + j * 16 + lm;
                const int bb = col >> 10, hw = col & 1023;
                float v = acc[i][j][r];
                if (alpha) v = v * alpha[row] + beta[row];
                else       v = v + beta[row];
                if (resid) v += resid[(long)bb * stRes + (long)row * kHW + hw];
                if (reluOut) v = fmaxf(v, 0.f);
                out[(long)bb * stOut + (long)row * kHW + hw] = v;
            }
}

// ---------------------------------------------------------------------------
// Elementwise / reduction helpers
// ---------------------------------------------------------------------------
// n8 groups of 8 elements
__global__ void cvt_f32_bf16(const float* __restrict__ s, bf16* __restrict__ d, long n8)
{
    long i = (long)blockIdx.x * blockDim.x + threadIdx.x;
    if (i >= n8) return;
    const float4 x0 = ((const float4*)s)[2 * i];
    const float4 x1 = ((const float4*)s)[2 * i + 1];
    v8bf t;
    t[0] = f2b(x0.x); t[1] = f2b(x0.y); t[2] = f2b(x0.z); t[3] = f2b(x0.w);
    t[4] = f2b(x1.x); t[5] = f2b(x1.y); t[6] = f2b(x1.z); t[7] = f2b(x1.w);
    ((v8bf*)d)[i] = t;
}

__global__ void bn_coeffs(const float* g, const float* bb, const float* m, const float* v,
                          const float* bias, float* alpha, float* beta, int n)
{
    int i = blockIdx.x * blockDim.x + threadIdx.x;
    if (i >= n) return;
    float s = g[i] * rsqrtf(v[i] + 1e-5f);
    alpha[i] = s;
    beta[i]  = bias[i] * s + bb[i] - m[i] * s;
}

// features (B,T,C,HW) frames {t0,t1,t2}[0:nf] -> bf16 keys (B,C,nf*HW); n4 = n/4
__global__ void gather_frames_bf16(const float* __restrict__ feat, int t0, int t1, int t2,
                                   int nf, bf16* __restrict__ dst, long n4)
{
    long i = (long)blockIdx.x * blockDim.x + threadIdx.x;
    if (i >= n4) return;
    const long i4 = i * 4;
    const int hw = (int)(i4 & 1023);
    long r = i4 >> 10;
    const int f = (int)(r % nf); r /= nf;
    const int c = (int)(r % kC);
    const int b = (int)(r / kC);
    const int t = (f == 0) ? t0 : (f == 1) ? t1 : t2;
    const float4 x = *(const float4*)&feat[(((long)b * kT + t) * kC + c) * kHW + hw];
    v4bf o;
    o[0] = f2b(x.x); o[1] = f2b(x.y); o[2] = f2b(x.z); o[3] = f2b(x.w);
    *(v4bf*)&dst[i4] = o;
}

// asq[b,k] = sum_c keys[b,c,k]^2
__global__ void colsumsq(const bf16* __restrict__ keys, float* __restrict__ asq, int Kd, long n)
{
    long i = (long)blockIdx.x * blockDim.x + threadIdx.x;
    if (i >= n) return;
    const int k = (int)(i % Kd);
    const int b = (int)(i / Kd);
    const bf16* p = keys + (long)b * kC * Kd + k;
    float s = 0.f;
    for (int c = 0; c < kC; ++c) { float v = (float)p[(long)c * Kd]; s += v * v; }
    asq[i] = s;
}

// in-place softmax over k of (2*ab - asq[k]) / sqrt(C), per (b,q) column
__global__ __launch_bounds__(256) void softmax_cols(float* __restrict__ logits,
                                                    const float* __restrict__ asq,
                                                    int Kd, int Q)
{
    __shared__ float s_asq[3072];
    const int b = blockIdx.y;
    const int q = blockIdx.x * 256 + threadIdx.x;
    const float inv = 0.04419417382415922f;   // 1/sqrt(512)
    for (int i = threadIdx.x; i < Kd; i += 256) s_asq[i] = asq[(long)b * Kd + i];
    __syncthreads();
    float* p = logits + (long)b * Kd * Q + q;
    float mx = -1e30f;
    for (int k = 0; k < Kd; ++k) {
        float t = (2.f * p[(long)k * Q] - s_asq[k]) * inv;
        mx = fmaxf(mx, t);
    }
    float sum = 0.f;
    for (int k = 0; k < Kd; ++k) {
        float t = __expf((2.f * p[(long)k * Q] - s_asq[k]) * inv - mx);
        p[(long)k * Q] = t;
        sum += t;
    }
    const float r = 1.f / sum;
    for (int k = 0; k < Kd; ++k) p[(long)k * Q] *= r;
}

// res = |l_aff - self_aff| (broadcast over Tl), downconvert to bf16
// NOTE: kept scalar — the float4 version triggers a gfx1250 ISel crash on the
// packed-f32 fabs idiom (v2f32 bitcast/and combine).
__global__ void residual_abs_bf16(const float* __restrict__ laff, const float* __restrict__ self,
                                  bf16* __restrict__ out, int Kl, long n)
{
    long i = (long)blockIdx.x * blockDim.x + threadIdx.x;
    if (i >= n) return;
    const int q = (int)(i & 1023);
    long r = i >> 10;
    const int k = (int)(r % Kl);
    const int b = (int)(r / Kl);
    const int k0 = k & 1023;
    out[i] = f2b(fabsf(laff[i] - self[((long)(b * 1024 + k0)) * 1024 + q]));
}

// short_fuse = features[:,idx] + short_read  (n4 = n/4)
__global__ void add_frame(const float* __restrict__ feat, int idx,
                          const float* __restrict__ srd, float* __restrict__ out, long n4)
{
    long i = (long)blockIdx.x * blockDim.x + threadIdx.x;
    if (i >= n4) return;
    const long i4 = i * 4;
    const long inner = i4 % kFrame;
    const long b     = i4 / kFrame;
    const float4 a = *(const float4*)&feat[b * kFeatB + (long)idx * kFrame + inner];
    const float4 s = *(const float4*)&srd[i4];
    float4 o = make_float4(a.x + s.x, a.y + s.y, a.z + s.z, a.w + s.w);
    *(float4*)&out[i4] = o;
}

// features[:,idx] = src  (n4 = n/4)
__global__ void scatter_frame(float* __restrict__ feat, int idx,
                              const float* __restrict__ src, long n4)
{
    long i = (long)blockIdx.x * blockDim.x + threadIdx.x;
    if (i >= n4) return;
    const long i4 = i * 4;
    const long inner = i4 % kFrame;
    const long b     = i4 / kFrame;
    *(float4*)&feat[b * kFeatB + (long)idx * kFrame + inner] = *(const float4*)&src[i4];
}

// ---------------------------------------------------------------------------
static inline long cdiv(long a, long b) { return (a + b - 1) / b; }

extern "C" void kernel_launch(void* const* d_in, const int* in_sizes, int n_in,
                              void* d_out, int out_size, void* d_ws, size_t ws_size,
                              hipStream_t stream)
{
    const float* f_in = (const float*)d_in[0];
    const float* W1[2]   = { (const float*)d_in[1],  (const float*)d_in[15] };
    const float* B1[2]   = { (const float*)d_in[2],  (const float*)d_in[16] };
    const float* W2[2]   = { (const float*)d_in[3],  (const float*)d_in[17] };
    const float* B2[2]   = { (const float*)d_in[4],  (const float*)d_in[18] };
    const float* WD[2]   = { (const float*)d_in[5],  (const float*)d_in[19] };
    const float* BD[2]   = { (const float*)d_in[6],  (const float*)d_in[20] };
    const float* G1[2]   = { (const float*)d_in[7],  (const float*)d_in[21] };
    const float* Bb1[2]  = { (const float*)d_in[8],  (const float*)d_in[22] };
    const float* M1[2]   = { (const float*)d_in[9],  (const float*)d_in[23] };
    const float* V1[2]   = { (const float*)d_in[10], (const float*)d_in[24] };
    const float* G2[2]   = { (const float*)d_in[11], (const float*)d_in[25] };
    const float* Bb2[2]  = { (const float*)d_in[12], (const float*)d_in[26] };
    const float* M2[2]   = { (const float*)d_in[13], (const float*)d_in[27] };
    const float* V2[2]   = { (const float*)d_in[14], (const float*)d_in[28] };
    const int CIN[2] = { 1024, 1536 };

    float* feat = (float*)d_out;
    (void)hipMemcpyAsync(feat, f_in, (size_t)kB * kFeatB * sizeof(float),
                         hipMemcpyDeviceToDevice, stream);

    // ---- workspace carve-out --------------------------------------------
    char* ws = (char*)d_ws;
    size_t off = 0;
    auto alloc = [&](size_t bytes) -> char* {
        char* p = ws + off;
        off += (bytes + 255) & ~(size_t)255;
        return p;
    };
    bf16* wbuf1[2]; bf16* wbuf2[2]; bf16* wbufd[2];
    float* al1[2]; float* be1[2]; float* al2[2]; float* be2[2];
    for (int s = 0; s < 2; ++s) {
        wbuf1[s] = (bf16*)alloc((size_t)512 * CIN[s] * 9 * sizeof(bf16));
        wbuf2[s] = (bf16*)alloc((size_t)512 * 512 * 9 * sizeof(bf16));
        wbufd[s] = (bf16*)alloc((size_t)512 * CIN[s] * 9 * sizeof(bf16));
        al1[s] = (float*)alloc(512 * 4); be1[s] = (float*)alloc(512 * 4);
        al2[s] = (float*)alloc(512 * 4); be2[s] = (float*)alloc(512 * 4);
    }
    bf16*  anchor  = (bf16*)alloc((size_t)kB * kC * kHW * 2);
    bf16*  keys_s  = (bf16*)alloc((size_t)kB * kC * 2048 * 2);
    bf16*  keys_l  = (bf16*)alloc((size_t)kB * kC * 3072 * 2);
    float* asq_a   = (float*)alloc((size_t)kB * 1024 * 4);
    float* asq_s   = (float*)alloc((size_t)kB * 2048 * 4);
    float* asq_l   = (float*)alloc((size_t)kB * 3072 * 4);
    float* ab_s    = (float*)alloc((size_t)kB * 2048 * 1024 * 4);
    bf16*  aff_s   = (bf16*)alloc((size_t)kB * 2048 * 1024 * 2);
    float* ab_l    = (float*)alloc((size_t)kB * 3072 * 1024 * 4);
    float* self_ab = (float*)alloc((size_t)kB * 1024 * 1024 * 4);
    bf16*  res_bf  = (bf16*)alloc((size_t)kB * 3072 * 1024 * 2);
    float* s_read  = (float*)alloc((size_t)kB * kFrame * 4);
    float* l_read  = (float*)alloc((size_t)kB * kFrame * 4);
    float* s_fuse  = (float*)alloc((size_t)kB * kFrame * 4);
    float* l_fuse  = (float*)alloc((size_t)kB * kFrame * 4);
    float* r1      = (float*)alloc((size_t)kB * kFrame * 4);
    float* r2      = (float*)alloc((size_t)kB * kFrame * 4);
    float* fus     = (float*)alloc((size_t)kB * kFrame * 4);

    // ---- one-time: weight downconvert + folded BN coefficients ----------
    for (int s = 0; s < 2; ++s) {
        long n1 = (long)512 * CIN[s] * 9 / 8, n2 = (long)512 * 512 * 9 / 8;
        cvt_f32_bf16<<<cdiv(n1, 256), 256, 0, stream>>>(W1[s], wbuf1[s], n1);
        cvt_f32_bf16<<<cdiv(n2, 256), 256, 0, stream>>>(W2[s], wbuf2[s], n2);
        cvt_f32_bf16<<<cdiv(n1, 256), 256, 0, stream>>>(WD[s], wbufd[s], n1);
        bn_coeffs<<<2, 256, 0, stream>>>(G1[s], Bb1[s], M1[s], V1[s], B1[s], al1[s], be1[s], 512);
        bn_coeffs<<<2, 256, 0, stream>>>(G2[s], Bb2[s], M2[s], V2[s], B2[s], al2[s], be2[s], 512);
    }

    // short/long frame index tables (T=5)
    const int sr0[5] = {0, 0, 1, 2, 3};
    const int sr2[5] = {1, 2, 3, 4, 4};
    const int lrA[5] = {2, 3, 0, 0, 0};
    const int lrB[5] = {3, 4, 4, 1, 1};
    const int lrC[5] = {4, 0, 0, 0, 2};
    const int TlT[5] = {3, 2, 2, 2, 3};

    const long nFr4 = (long)kB * kFrame / 4;
    const dim3 blk(256);

    for (int idx = 0; idx < kT; ++idx) {
        const int Tl = TlT[idx];
        const int Kl = Tl * kHW;                 // 2048 or 3072
        const float* cur = feat + (long)idx * kFrame;   // batch stride kFeatB

        long nA4 = (long)kB * kC * kHW / 4;
        long nS4 = (long)kB * kC * 2048 / 4;
        long nL4 = (long)kB * kC * Kl / 4;
        gather_frames_bf16<<<cdiv(nA4, 256), blk, 0, stream>>>(feat, idx, 0, 0, 1, anchor, nA4);
        gather_frames_bf16<<<cdiv(nS4, 256), blk, 0, stream>>>(feat, sr0[idx], sr2[idx], 0, 2, keys_s, nS4);
        gather_frames_bf16<<<cdiv(nL4, 256), blk, 0, stream>>>(feat, lrA[idx], lrB[idx], lrC[idx], Tl, keys_l, nL4);

        colsumsq<<<cdiv((long)kB * 1024, 256), blk, 0, stream>>>(anchor, asq_a, 1024, (long)kB * 1024);
        colsumsq<<<cdiv((long)kB * 2048, 256), blk, 0, stream>>>(keys_s, asq_s, 2048, (long)kB * 2048);
        colsumsq<<<cdiv((long)kB * Kl,  256), blk, 0, stream>>>(keys_l, asq_l, Kl,  (long)kB * Kl);

        // ab = keys^T @ anchor  (TN, K=C=512)
        gemm_bf16<true><<<dim3(2048 / 64, 1024 / 128, kB), blk, 0, stream>>>(
            keys_s, anchor, ab_s, 2048, 1024, kC,
            (long)kC * 2048, (long)kC * 1024, (long)2048 * 1024);
        gemm_bf16<true><<<dim3(Kl / 64, 1024 / 128, kB), blk, 0, stream>>>(
            keys_l, anchor, ab_l, Kl, 1024, kC,
            (long)kC * Kl, (long)kC * 1024, (long)Kl * 1024);
        gemm_bf16<true><<<dim3(1024 / 64, 1024 / 128, kB), blk, 0, stream>>>(
            anchor, anchor, self_ab, 1024, 1024, kC,
            (long)kC * 1024, (long)kC * 1024, (long)1024 * 1024);

        softmax_cols<<<dim3(1024 / 256, kB), blk, 0, stream>>>(ab_s, asq_s, 2048, 1024);
        softmax_cols<<<dim3(1024 / 256, kB), blk, 0, stream>>>(ab_l, asq_l, Kl, 1024);
        softmax_cols<<<dim3(1024 / 256, kB), blk, 0, stream>>>(self_ab, asq_a, 1024, 1024);

        long nAffS8 = (long)kB * 2048 * 1024 / 8;
        cvt_f32_bf16<<<cdiv(nAffS8, 256), blk, 0, stream>>>(ab_s, aff_s, nAffS8);
        long nRes = (long)kB * Kl * 1024;
        residual_abs_bf16<<<cdiv(nRes, 256), blk, 0, stream>>>(ab_l, self_ab, res_bf, Kl, nRes);

        // reads: keys @ aff  (NN, M=C)
        gemm_bf16<false><<<dim3(kC / 64, 1024 / 128, kB), blk, 0, stream>>>(
            keys_s, aff_s, s_read, kC, 1024, 2048,
            (long)kC * 2048, (long)2048 * 1024, kFrame);
        gemm_bf16<false><<<dim3(kC / 64, 1024 / 128, kB), blk, 0, stream>>>(
            keys_l, res_bf, l_read, kC, 1024, Kl,
            (long)kC * Kl, (long)Kl * 1024, kFrame);

        add_frame<<<cdiv(nFr4, 256), blk, 0, stream>>>(feat, idx, s_read, s_fuse, nFr4);

        const dim3 gconv(512 / 64, kN / 128, 1);

        // ---- fuse1 lg: x = [cur, long_read] (Cin=1024) -------------------
        conv_gemm_bf16<<<gconv, blk, 0, stream>>>(
            wbuf1[0], cur, kFeatB, l_read, kFrame, nullptr, 0,
            1024, 1, r1, kFrame, al1[0], be1[0], nullptr, 0, 1);
        conv_gemm_bf16<<<gconv, blk, 0, stream>>>(
            wbuf2[0], r1, kFrame, nullptr, 0, nullptr, 0,
            512, 0, r2, kFrame, al2[0], be2[0], nullptr, 0, 0);
        conv_gemm_bf16<<<gconv, blk, 0, stream>>>(
            wbufd[0], cur, kFeatB, l_read, kFrame, nullptr, 0,
            1024, 0, l_fuse, kFrame, nullptr, BD[0], r2, kFrame, 1);

        // ---- fuse1 ig: x = [cur, short_fuse, long_fuse] (Cin=1536) ------
        conv_gemm_bf16<<<gconv, blk, 0, stream>>>(
            wbuf1[1], cur, kFeatB, s_fuse, kFrame, l_fuse, kFrame,
            1536, 1, r1, kFrame, al1[1], be1[1], nullptr, 0, 1);
        conv_gemm_bf16<<<gconv, blk, 0, stream>>>(
            wbuf2[1], r1, kFrame, nullptr, 0, nullptr, 0,
            512, 0, r2, kFrame, al2[1], be2[1], nullptr, 0, 0);
        conv_gemm_bf16<<<gconv, blk, 0, stream>>>(
            wbufd[1], cur, kFeatB, s_fuse, kFrame, l_fuse, kFrame,
            1536, 0, fus, kFrame, nullptr, BD[1], r2, kFrame, 1);

        scatter_frame<<<cdiv(nFr4, 256), blk, 0, stream>>>(feat, idx, fus, nFr4);
    }
}
